// MultiHeadAttention_34041910788859
// MI455X (gfx1250) — matrix operand order repaired and checked by reading, not compiled, not run on hardware
//
#include <hip/hip_runtime.h>

// ---------------------------------------------------------------------------
// MHA for MI455X (gfx1250, wave32, WMMA f16 / f32-accum).
//   1) f32->f16 pre-convert of X (q/k/v inputs) and W (q/k/v/o weights)
//   2) q/k/v projections: f16 WMMA GEMM (score scale 1/8 folded into q)
//   3) flash attention over [G=32, S=2048, dk=64] flat view, 64-key blocks,
//      DPP-based online softmax, async LDS staging of the K tile
//   4) output projection -> f32
// Workspace: 32M halves = 64 MB.
// ---------------------------------------------------------------------------

typedef __attribute__((ext_vector_type(16))) _Float16 v16h;
typedef __attribute__((ext_vector_type(8)))  float    v8f;
typedef __attribute__((ext_vector_type(4)))  int      v4i;

#define MHA_B   2
#define MHA_S   2048
#define MHA_D   1024
#define MHA_G   32
#define MHA_DK  64
#define MHA_M   (MHA_B * MHA_S)
#define KB      64              // keys per attention block iteration

#if defined(__has_builtin)
#  if __has_builtin(__builtin_amdgcn_global_load_async_to_lds_b128) && \
      __has_builtin(__builtin_amdgcn_s_wait_asynccnt)
#    define ATT_ASYNC_STAGE 1
#  endif
#  if __has_builtin(__builtin_amdgcn_update_dpp)
#    define ATT_DPP 1
#  endif
#endif
#ifndef ATT_ASYNC_STAGE
#  define ATT_ASYNC_STAGE 0
#endif
#ifndef ATT_DPP
#  define ATT_DPP 0
#endif

union HalfVec16 {
  v16h     v;
  _Float16 h[16];
  uint4    q[2];
};

__device__ __forceinline__ v8f wmma_f16(v16h a, v16h b, v8f c) {
  return __builtin_amdgcn_wmma_f32_16x16x32_f16(
      false, a, false, b, (short)0, c, false, false);
}

// A-fragment (16x32 f16): lane l holds row m = l%16; element e -> k =
// k0 + (e/8)*16 + (e%8); caller passes k0 = kbase + (lane>=16 ? 8 : 0).
__device__ __forceinline__ v16h load_a_f16(const _Float16* base, int ld, int m, int k0) {
  HalfVec16 r;
  const _Float16* p = base + (size_t)m * ld + k0;
  r.q[0] = *(const uint4*)(p);
  r.q[1] = *(const uint4*)(p + 16);
  return r.v;
}

// B-fragment (32x16 f16): lane l holds col n = l%16; element e -> k =
// (lane>=16 ? 8 : 0) + (e/8)*16 + (e%8); pointer is to the lane's first half.
__device__ __forceinline__ v16h load_b_f16(const _Float16* p) {
  HalfVec16 r;
  r.q[0] = ((const uint4*)p)[0];
  r.q[1] = ((const uint4*)p)[2];
  return r.v;
}

// ---- 16-lane-half reductions (wave32): DPP if available, else shfl --------
#if ATT_DPP
template <int CTRL>
__device__ __forceinline__ float dpp_movf(float x) {
  int xi = __builtin_bit_cast(int, x);
  return __builtin_bit_cast(float,
      __builtin_amdgcn_update_dpp(xi, xi, CTRL, 0xF, 0xF, true));
}
__device__ __forceinline__ float red16_max(float x) {
  x = fmaxf(x, dpp_movf<0xB1>(x));    // quad_perm(1,0,3,2): xor1
  x = fmaxf(x, dpp_movf<0x4E>(x));    // quad_perm(2,3,0,1): xor2
  x = fmaxf(x, dpp_movf<0x141>(x));   // row_half_mirror -> 8-wide
  x = fmaxf(x, dpp_movf<0x140>(x));   // row_mirror      -> 16-wide
  return x;
}
__device__ __forceinline__ float red16_sum(float x) {
  x += dpp_movf<0xB1>(x);
  x += dpp_movf<0x4E>(x);
  x += dpp_movf<0x141>(x);
  x += dpp_movf<0x140>(x);
  return x;
}
#else
__device__ __forceinline__ float red16_max(float x) {
#pragma unroll
  for (int m = 1; m < 16; m <<= 1) x = fmaxf(x, __shfl_xor(x, m, 32));
  return x;
}
__device__ __forceinline__ float red16_sum(float x) {
#pragma unroll
  for (int m = 1; m < 16; m <<= 1) x += __shfl_xor(x, m, 32);
  return x;
}
#endif

// ---------------------------------------------------------------------------
// Elementwise f32 -> f16 (8 elems/thread; n is a multiple of 2048).
// ---------------------------------------------------------------------------
__global__ __launch_bounds__(256)
void cvt_f32_to_f16_kernel(const float* __restrict__ src,
                           _Float16* __restrict__ dst, int n) {
  const int i = (blockIdx.x * 256 + threadIdx.x) * 8;
  if (i + 8 > n) return;
  float4 f0 = *(const float4*)(src + i);
  float4 f1 = *(const float4*)(src + i + 4);
  union { _Float16 h[8]; uint4 q; } r;
  r.h[0] = (_Float16)f0.x; r.h[1] = (_Float16)f0.y;
  r.h[2] = (_Float16)f0.z; r.h[3] = (_Float16)f0.w;
  r.h[4] = (_Float16)f1.x; r.h[5] = (_Float16)f1.y;
  r.h[6] = (_Float16)f1.z; r.h[7] = (_Float16)f1.w;
  *(uint4*)(dst + i) = r.q;
}

// ---------------------------------------------------------------------------
// GEMM: Out[M,N] = (A[M,K] @ W[N,K]^T + bias[N]) * out_scale, all-f16 inputs.
// One wave computes a 32(M) x 64(N) tile; 8 wmma per 32-wide K step.
// ---------------------------------------------------------------------------
template <bool OUT_F16>
__global__ __launch_bounds__(32)
void gemm_f16_kernel(const _Float16* __restrict__ A, const _Float16* __restrict__ Wt,
                     const float* __restrict__ bias, void* __restrict__ Out,
                     int M, int N, int K, float out_scale) {
  const int lane  = threadIdx.x & 31;
  const int lm    = lane & 15;
  const bool hi   = lane >= 16;
  const int ntile = blockIdx.x * 64;
  const int mtile = blockIdx.y * 32;

  v8f zero = {};
  v8f acc0[4] = {zero, zero, zero, zero};
  v8f acc1[4] = {zero, zero, zero, zero};

  for (int kk = 0; kk < K; kk += 32) {
    const int ka = kk + (hi ? 8 : 0);
    v16h a0 = load_a_f16(A, K, mtile + lm, ka);
    v16h a1 = load_a_f16(A, K, mtile + 16 + lm, ka);
    const int kb = kk + (hi ? 8 : 0);
#pragma unroll
    for (int j = 0; j < 4; ++j) {
      v16h b = load_b_f16(Wt + (size_t)(ntile + j * 16 + lm) * K + kb);
      acc0[j] = wmma_f16(a0, b, acc0[j]);
      acc1[j] = wmma_f16(a1, b, acc1[j]);
    }
  }

#pragma unroll
  for (int j = 0; j < 4; ++j) {
    const int n = ntile + j * 16 + lm;
    const float bj = bias[n];
#pragma unroll
    for (int r = 0; r < 8; ++r) {
      const int m0 = mtile + r + (hi ? 8 : 0);
      const float v0 = (acc0[j][r] + bj) * out_scale;
      const float v1 = (acc1[j][r] + bj) * out_scale;
      if (OUT_F16) {
        ((_Float16*)Out)[(size_t)m0 * N + n]        = (_Float16)v0;
        ((_Float16*)Out)[(size_t)(m0 + 16) * N + n] = (_Float16)v1;
      } else {
        ((float*)Out)[(size_t)m0 * N + n]        = v0;
        ((float*)Out)[(size_t)(m0 + 16) * N + n] = v1;
      }
    }
  }
}

// ---------------------------------------------------------------------------
// Flash attention, 64-key blocks. Block = 128 threads = 4 waves, each wave a
// 16-query tile; K tile staged (async if available), V tile staged transposed.
// Scale (1/8) already folded into q. 16 wmma per key block per wave.
// ---------------------------------------------------------------------------
__global__ __launch_bounds__(128)
void flash_attention_kernel(const _Float16* __restrict__ q,
                            const _Float16* __restrict__ k,
                            const _Float16* __restrict__ v,
                            _Float16* __restrict__ out) {
  __shared__ _Float16 kbuf[KB * MHA_DK];        // [key][d]   8 KB
  __shared__ _Float16 vbufT[MHA_DK * KB];       // [d][key]   8 KB
  __shared__ _Float16 pbuf[4][16 * KB];         // per-wave P 2 KB each

  const int tid   = threadIdx.x;
  const int wave  = tid >> 5;
  const int lane  = tid & 31;
  const int lm    = lane & 15;
  const bool hi   = lane >= 16;
  const int g     = blockIdx.y;
  const int qbase = blockIdx.x * 64 + wave * 16;
  const size_t gbase = (size_t)g * MHA_S * MHA_DK;

  const v16h aq0 = load_a_f16(q + gbase, MHA_DK, qbase + lm, (hi ? 8 : 0));
  const v16h aq1 = load_a_f16(q + gbase, MHA_DK, qbase + lm, 32 + (hi ? 8 : 0));

  v8f zero = {};
  v8f o[4] = {zero, zero, zero, zero};
  float rmax[8], rsum[8];
#pragma unroll
  for (int r = 0; r < 8; ++r) { rmax[r] = -1e30f; rsum[r] = 0.0f; }

  for (int kblk = 0; kblk < MHA_S / KB; ++kblk) {
    const _Float16* ksrc = k + gbase + (size_t)kblk * KB * MHA_DK;
    const _Float16* vsrc = v + gbase + (size_t)kblk * KB * MHA_DK;

    __syncthreads();
    // ---- stage K tile: 4096 halves, 32 per thread ----
    {
      const int off = tid * 32;
#if ATT_ASYNC_STAGE
#pragma unroll
      for (int i = 0; i < 4; ++i)
        __builtin_amdgcn_global_load_async_to_lds_b128(
            (v4i*)(ksrc + off + i * 8),
            (v4i*)(&kbuf[off + i * 8]), 0, 0);
#else
#pragma unroll
      for (int i = 0; i < 4; ++i)
        *(uint4*)&kbuf[off + i * 8] = *(const uint4*)(ksrc + off + i * 8);
#endif
    }
    // ---- stage V tile transposed: thread = 2 keys x 16 d, b32 LDS stores ----
    {
      const int kp = tid >> 2;               // key pair 0..31
      const int d0 = (tid & 3) * 16;         // 0,16,32,48
      HalfVec16 r0, r1;
      r0.q[0] = *(const uint4*)(vsrc + (2 * kp) * MHA_DK + d0);
      r0.q[1] = *(const uint4*)(vsrc + (2 * kp) * MHA_DK + d0 + 8);
      r1.q[0] = *(const uint4*)(vsrc + (2 * kp + 1) * MHA_DK + d0);
      r1.q[1] = *(const uint4*)(vsrc + (2 * kp + 1) * MHA_DK + d0 + 8);
#pragma unroll
      for (int i = 0; i < 16; ++i) {
        union { _Float16 h[2]; unsigned u; } pk;
        pk.h[0] = r0.h[i]; pk.h[1] = r1.h[i];
        *(unsigned*)&vbufT[(d0 + i) * KB + 2 * kp] = pk.u;
      }
      if (kblk + 1 < MHA_S / KB) {
        __builtin_prefetch(ksrc + KB * MHA_DK + tid * 32, 0, 1);
        __builtin_prefetch(vsrc + KB * MHA_DK + tid * 32, 0, 1);
      }
    }
#if ATT_ASYNC_STAGE
    __builtin_amdgcn_s_wait_asynccnt(0);
#endif
    __syncthreads();

    // ---- scores: 4 tiles of 16 keys (8 wmma) ----
    v8f s[4];
#pragma unroll
    for (int nb = 0; nb < 4; ++nb) {
      const _Float16* kp0 = &kbuf[(nb * 16 + lm) * MHA_DK + (hi ? 8 : 0)];
      v16h b0 = load_b_f16(kp0);
      v16h b1 = load_b_f16(kp0 + 32);
      v8f z = zero;
      z     = wmma_f16(aq0, b0, z);
      s[nb] = wmma_f16(aq1, b1, z);
    }

    // ---- online softmax over 64 new columns ----
#pragma unroll
    for (int r = 0; r < 8; ++r) {
      const float x0 = s[0][r], x1 = s[1][r], x2 = s[2][r], x3 = s[3][r];
      float mx = red16_max(fmaxf(fmaxf(x0, x1), fmaxf(x2, x3)));
      const float nm   = fmaxf(rmax[r], mx);
      const float corr = __expf(rmax[r] - nm);
      const float p0 = __expf(x0 - nm), p1 = __expf(x1 - nm);
      const float p2 = __expf(x2 - nm), p3 = __expf(x3 - nm);
      const float ps = red16_sum((p0 + p1) + (p2 + p3));
      rsum[r] = rsum[r] * corr + ps;
      rmax[r] = nm;
#pragma unroll
      for (int j = 0; j < 4; ++j) o[j][r] *= corr;

      _Float16* pr = &pbuf[wave][(r + (hi ? 8 : 0)) * KB];
      pr[lm]      = (_Float16)p0;
      pr[16 + lm] = (_Float16)p1;
      pr[32 + lm] = (_Float16)p2;
      pr[48 + lm] = (_Float16)p3;
    }

    // ---- P (16x64) as two A-frags; PV (8 wmma) ----
    v16h ap0 = load_a_f16(&pbuf[wave][0], KB, lm, (hi ? 8 : 0));
    v16h ap1 = load_a_f16(&pbuf[wave][0], KB, lm, 32 + (hi ? 8 : 0));
#pragma unroll
    for (int j = 0; j < 4; ++j) {
      const _Float16* vp = &vbufT[(j * 16 + lm) * KB + (hi ? 8 : 0)];
      v16h bv0 = load_b_f16(vp);
      v16h bv1 = load_b_f16(vp + 32);
      o[j] = wmma_f16(ap0, bv0, o[j]);
      o[j] = wmma_f16(ap1, bv1, o[j]);
    }
  }

  // ---- normalize + store attended (f16, flat [B,S,D] view) ----
#pragma unroll
  for (int j = 0; j < 4; ++j) {
#pragma unroll
    for (int r = 0; r < 8; ++r) {
      const int m = qbase + r + (hi ? 8 : 0);
      const float inv = 1.0f / rsum[r];
      out[gbase + (size_t)m * MHA_DK + j * 16 + lm] = (_Float16)(o[j][r] * inv);
    }
  }
}

// ---------------------------------------------------------------------------
extern "C" void kernel_launch(void* const* d_in, const int* in_sizes, int n_in,
                              void* d_out, int out_size, void* d_ws, size_t ws_size,
                              hipStream_t stream) {
  const float* query = (const float*)d_in[0];
  const float* key_  = (const float*)d_in[1];
  const float* value = (const float*)d_in[2];
  const float* Wq    = (const float*)d_in[3];
  const float* bq    = (const float*)d_in[4];
  const float* Wk    = (const float*)d_in[5];
  const float* bk    = (const float*)d_in[6];
  const float* Wv    = (const float*)d_in[7];
  const float* bv    = (const float*)d_in[8];
  const float* Wo    = (const float*)d_in[9];
  const float* bo    = (const float*)d_in[10];

  const int M = MHA_M, N = MHA_D, K = MHA_D;
  const size_t MN = (size_t)M * N;   // 4M elements
  const size_t NK = (size_t)N * K;   // 1M elements

  // Workspace layout (f16): 3*MN (x) + 4*NK (w) + 4*MN (q/k/v/att) = 64 MB.
  _Float16* p    = (_Float16*)d_ws;
  _Float16* xq16 = p; p += MN;
  _Float16* xk16 = p; p += MN;
  _Float16* xv16 = p; p += MN;
  _Float16* wq16 = p; p += NK;
  _Float16* wk16 = p; p += NK;
  _Float16* wv16 = p; p += NK;
  _Float16* wo16 = p; p += NK;
  _Float16* q16  = p; p += MN;
  _Float16* k16  = p; p += MN;
  _Float16* v16  = p; p += MN;
  _Float16* a16  = p; p += MN;

  auto cvt = [&](const float* s, _Float16* d, size_t n) {
    cvt_f32_to_f16_kernel<<<dim3((unsigned)(n / 8 / 256)), 256, 0, stream>>>(s, d, (int)n);
  };
  cvt(query, xq16, MN);
  cvt(key_,  xk16, MN);
  cvt(value, xv16, MN);
  cvt(Wq, wq16, NK);
  cvt(Wk, wk16, NK);
  cvt(Wv, wv16, NK);
  cvt(Wo, wo16, NK);

  dim3 ggrid(N / 64, M / 32);   // (16, 128)
  // Score scale 1/sqrt(dk)=0.125 folded into the q projection.
  gemm_f16_kernel<true><<<ggrid, 32, 0, stream>>>(xq16, wq16, bq, q16, M, N, K, 0.125f);
  gemm_f16_kernel<true><<<ggrid, 32, 0, stream>>>(xk16, wk16, bk, k16, M, N, K, 1.0f);
  gemm_f16_kernel<true><<<ggrid, 32, 0, stream>>>(xv16, wv16, bv, v16, M, N, K, 1.0f);

  flash_attention_kernel<<<dim3(MHA_S / 64, MHA_G), 128, 0, stream>>>(q16, k16, v16, a16);

  gemm_f16_kernel<false><<<ggrid, 32, 0, stream>>>(a16, wo16, bo, d_out, M, N, K, 1.0f);
}
